// TinyAttention_19009525252535
// MI455X (gfx1250) — compile-verified, matrix operand order checked by reading, not run on hardware
//
#include <hip/hip_runtime.h>

typedef __attribute__((ext_vector_type(16))) __bf16 v16bf;
typedef __attribute__((ext_vector_type(8)))  float  v8f;

#define LDS_STRIDE 40  // 32 + 8 pad elements; rows stay 16B-aligned

__device__ inline __bf16 f32_to_bf16(float f) {
  unsigned u = __builtin_bit_cast(unsigned, f);
  unsigned r = u + 0x7FFFu + ((u >> 16) & 1u);   // round-to-nearest-even
  unsigned short h = (unsigned short)(r >> 16);
  return __builtin_bit_cast(__bf16, h);
}
__device__ inline float bf16_to_f32(__bf16 b) {
  unsigned short h = __builtin_bit_cast(unsigned short, b);
  unsigned u = ((unsigned)h) << 16;
  return __builtin_bit_cast(float, u);
}
__device__ inline __bf16 to_bf16(float f)  { return f32_to_bf16(f); }
__device__ inline __bf16 to_bf16(__bf16 b) { return b; }

// ---- CDNA5 async global->LDS copy (ASYNCcnt-tracked, no VGPR round trip) ----
__device__ inline void gld_async_b128(unsigned ldsOff, const void* gp) {
  asm volatile("global_load_async_to_lds_b128 %0, %1, off"
               :: "v"(ldsOff), "v"(gp) : "memory");
}
__device__ inline void wait_async0() {
  asm volatile("s_wait_asynccnt 0" ::: "memory");
}

// ---------------- fp32 -> bf16 convert ----------------
__global__ __launch_bounds__(256)
void cvt_f32_bf16(const float* __restrict__ in, __bf16* __restrict__ out, long long n) {
  long long i = (long long)blockIdx.x * blockDim.x + threadIdx.x;
  long long stride = (long long)gridDim.x * blockDim.x;
  for (; i < n; i += stride) out[i] = f32_to_bf16(in[i]);
}

// ---------------- tiled transpose (f32 or bf16 in) -> bf16 out --------------
// out[z][c * ldo + r] = bf16(in[z][r * ldi + c]);  grid: (C/32, R/32, batches)
template <typename TIN>
__global__ __launch_bounds__(256)
void transpose_bf16(const TIN* __restrict__ in, __bf16* __restrict__ out,
                    int ldi, int ldo, long long sIn, long long sOut) {
  __shared__ __bf16 tile[32][33];
  const int tx = threadIdx.x, ty = threadIdx.y;  // blockDim = (32, 8)
  const long long zi = (long long)blockIdx.z * sIn;
  const long long zo = (long long)blockIdx.z * sOut;
  const int c0 = blockIdx.x * 32, r0 = blockIdx.y * 32;
#pragma unroll
  for (int i = 0; i < 4; ++i) {
    int r = r0 + ty + i * 8;
    tile[ty + i * 8][tx] = to_bf16(in[zi + (long long)r * ldi + c0 + tx]);
  }
  __syncthreads();
#pragma unroll
  for (int i = 0; i < 4; ++i) {
    int r = c0 + ty + i * 8;
    out[zo + (long long)r * ldo + r0 + tx] = tile[tx][ty + i * 8];
  }
}

// Load a 16x32 bf16 fragment from an LDS tile laid out [row][k], matching the
// CDNA5 16-bit A/B operand lane layout:
//   lanes 0-15 : row=lane,    elems 0-7 -> K 0-7,  elems 8-15 -> K 16-23
//   lanes 16-31: row=lane-16, elems 0-7 -> K 8-15, elems 8-15 -> K 24-31
__device__ inline v16bf load_frag(const __bf16* tile, int row0) {
  int lane = threadIdx.x & 31;
  int r = row0 + (lane & 15);
  int kh = (lane & 16) ? 8 : 0;
  const __bf16* p = tile + r * LDS_STRIDE + kh;
  v16bf f;
#pragma unroll
  for (int i = 0; i < 8; ++i) f[i] = p[i];
#pragma unroll
  for (int i = 0; i < 8; ++i) f[8 + i] = p[16 + i];
  return f;
}

// ---------------- WMMA GEMM: C[z] = A[z] (MxK, lda) @ B[z]^T (B is [N,K], ldb)
// EPI: 0 = store bf16; 1 = scale + causal mask -> bf16; 2 = f32 + bias.
// Double-buffered LDS fed by async global->LDS B128 copies.
template <int EPI>
__global__ __launch_bounds__(128)
void gemm_wmma(const __bf16* __restrict__ A, const __bf16* __restrict__ B,
               void* __restrict__ Cout, const float* __restrict__ bias,
               int K, int lda, int ldb, int ldc,
               long long sA, long long sB, long long sC, float scale) {
  __shared__ __align__(16) __bf16 ldsA[2][64 * LDS_STRIDE];
  __shared__ __align__(16) __bf16 ldsB[2][64 * LDS_STRIDE];

  const int tid  = threadIdx.x;
  const int lane = tid & 31;
  const int wave = tid >> 5;
  const int m0 = blockIdx.y * 64;
  const int n0 = blockIdx.x * 64;
  const int z  = blockIdx.z;
  A += (long long)z * sA;
  B += (long long)z * sB;

  const int wm = (wave >> 1) * 32;
  const int wn = (wave & 1) * 32;

  v8f zero = {0.f, 0.f, 0.f, 0.f, 0.f, 0.f, 0.f, 0.f};
  v8f c[2][2];
  c[0][0] = zero; c[0][1] = zero; c[1][0] = zero; c[1][1] = zero;

  // Blocks entirely above the causal diagonal: skip all math (block-uniform).
  const bool full_mask = (EPI == 1) && (n0 > m0 + 63);

  if (!full_mask) {
    // Per-thread copy slice: rows (tid>>2) and (tid>>2)+32, k-chunk (tid&3)*8.
    const int rowT = tid >> 2;
    const int colk = (tid & 3) * 8;
    const __bf16* gA = A + (long long)(m0 + rowT) * lda + colk;
    const __bf16* gB = B + (long long)(n0 + rowT) * ldb + colk;
    const long long a32 = 32LL * lda, b32 = 32LL * ldb;

    const unsigned laBase = (unsigned)(size_t)&ldsA[0][rowT * LDS_STRIDE + colk];
    const unsigned lbBase = (unsigned)(size_t)&ldsB[0][rowT * LDS_STRIDE + colk];
    const unsigned rowHop = 32u * LDS_STRIDE * 2u;       // +32 rows, bytes
    const unsigned bufHop = 64u * LDS_STRIDE * 2u;       // buffer 1 offset

    // Prologue: async-copy tile 0 into buffer 0.
    gld_async_b128(laBase,          gA);
    gld_async_b128(laBase + rowHop, gA + a32);
    gld_async_b128(lbBase,          gB);
    gld_async_b128(lbBase + rowHop, gB + b32);
    wait_async0();
    __syncthreads();

    int buf = 0;
    for (int k0 = 0; k0 < K; k0 += 32) {
      const int nbuf = buf ^ 1;
      if (k0 + 32 < K) {  // async-copy next tile while we compute this one
        const unsigned bo = nbuf ? bufHop : 0u;
        gld_async_b128(laBase + bo,          gA + k0 + 32);
        gld_async_b128(laBase + bo + rowHop, gA + a32 + k0 + 32);
        gld_async_b128(lbBase + bo,          gB + k0 + 32);
        gld_async_b128(lbBase + bo + rowHop, gB + b32 + k0 + 32);
      }

      v16bf a0 = load_frag(ldsA[buf], wm + 0);
      v16bf a1 = load_frag(ldsA[buf], wm + 16);
      v16bf b0 = load_frag(ldsB[buf], wn + 0);
      v16bf b1 = load_frag(ldsB[buf], wn + 16);

      c[0][0] = __builtin_amdgcn_wmma_f32_16x16x32_bf16(false, a0, false, b0,
                                                        (short)0, c[0][0], false, false);
      c[0][1] = __builtin_amdgcn_wmma_f32_16x16x32_bf16(false, a0, false, b1,
                                                        (short)0, c[0][1], false, false);
      c[1][0] = __builtin_amdgcn_wmma_f32_16x16x32_bf16(false, a1, false, b0,
                                                        (short)0, c[1][0], false, false);
      c[1][1] = __builtin_amdgcn_wmma_f32_16x16x32_bf16(false, a1, false, b1,
                                                        (short)0, c[1][1], false, false);

      wait_async0();        // next buffer fully resident (no-op on last iter)
      __syncthreads();      // all waves done reading `buf`, next buffer visible
      buf = nbuf;
    }
  }

  // ---- epilogue: C layout -> element (m,n): lane = n | (m>=8)<<4, vgpr = m&7
  const int nl = lane & 15;
  const int mb = (lane & 16) ? 8 : 0;
#pragma unroll
  for (int i = 0; i < 2; ++i) {
#pragma unroll
    for (int j = 0; j < 2; ++j) {
      int mrow = m0 + wm + i * 16 + mb;
      int ncol = n0 + wn + j * 16 + nl;
#pragma unroll
      for (int r = 0; r < 8; ++r) {
        float v = c[i][j][r];
        long long idx = (long long)z * sC + (long long)(mrow + r) * ldc + ncol;
        if (EPI == 0) {
          ((__bf16*)Cout)[idx] = f32_to_bf16(v);
        } else if (EPI == 1) {
          v *= scale;
          if (ncol > mrow + r) v = -1e30f;  // causal mask
          ((__bf16*)Cout)[idx] = f32_to_bf16(v);
        } else {
          ((float*)Cout)[idx] = v + bias[ncol];
        }
      }
    }
  }
}

// ---------------- in-place causal softmax over bf16 rows ----------------
__global__ __launch_bounds__(256)
void softmax_rows(__bf16* __restrict__ P, int T) {
  __bf16* row = P + ((long long)blockIdx.y * T + blockIdx.x) * (long long)T;
  const int tid = threadIdx.x;
  __shared__ float red[8];

  float vals[8];
  float mx = -1e30f;
#pragma unroll
  for (int i = 0; i < 8; ++i) {
    vals[i] = bf16_to_f32(row[tid + i * 256]);
    mx = fmaxf(mx, vals[i]);
  }
  for (int o = 16; o > 0; o >>= 1) mx = fmaxf(mx, __shfl_xor(mx, o, 32));
  if ((tid & 31) == 0) red[tid >> 5] = mx;
  __syncthreads();
  float bm = red[0];
#pragma unroll
  for (int w = 1; w < 8; ++w) bm = fmaxf(bm, red[w]);

  float sum = 0.f;
#pragma unroll
  for (int i = 0; i < 8; ++i) { vals[i] = __expf(vals[i] - bm); sum += vals[i]; }
  for (int o = 16; o > 0; o >>= 1) sum += __shfl_xor(sum, o, 32);
  __syncthreads();
  if ((tid & 31) == 0) red[tid >> 5] = sum;
  __syncthreads();
  float bs = 0.f;
#pragma unroll
  for (int w = 0; w < 8; ++w) bs += red[w];
  float inv = 1.0f / bs;
#pragma unroll
  for (int i = 0; i < 8; ++i) row[tid + i * 256] = f32_to_bf16(vals[i] * inv);
}

extern "C" void kernel_launch(void* const* d_in, const int* in_sizes, int n_in,
                              void* d_out, int out_size, void* d_ws, size_t ws_size,
                              hipStream_t stream) {
  (void)in_sizes; (void)n_in; (void)out_size; (void)ws_size;
  const float* x      = (const float*)d_in[0];  // [4,2048,1024]
  const float* w_qkv  = (const float*)d_in[1];  // [1024,3072]
  const float* w_proj = (const float*)d_in[2];  // [1024,1024]
  const float* b_proj = (const float*)d_in[3];  // [1024]
  float* y = (float*)d_out;                     // [4,2048,1024] fp32

  const int Bb = 4, T = 2048, D = 1024;
  const long long xN = (long long)Bb * T * D;

  char* ws = (char*)d_ws;
  __bf16* xb     = (__bf16*)ws; ws += xN * 2;                       // [8192,1024]
  __bf16* wqkvT  = (__bf16*)ws; ws += (long long)3 * D * D * 2;     // [3072,1024]
  __bf16* wprojT = (__bf16*)ws; ws += (long long)D * D * 2;         // [1024,1024]
  __bf16* qkvb   = (__bf16*)ws; ws += (long long)Bb * T * 3 * D * 2;// [b,t,3d]
  __bf16* vT     = (__bf16*)ws; ws += xN * 2;                       // [b,d,t]
  __bf16* pb     = (__bf16*)ws; ws += (long long)Bb * T * T * 2;    // [b,t,t]
  __bf16* outb   = (__bf16*)ws; ws += xN * 2;                       // [b,t,d]

  cvt_f32_bf16<<<1024, 256, 0, stream>>>(x, xb, xN);
  // w_qkv [1024,3072] -> wqkvT [3072,1024]
  transpose_bf16<float><<<dim3(3 * D / 32, D / 32, 1), dim3(32, 8), 0, stream>>>(
      w_qkv, wqkvT, 3 * D, D, 0, 0);
  // w_proj [1024,1024] -> wprojT [1024,1024]
  transpose_bf16<float><<<dim3(D / 32, D / 32, 1), dim3(32, 8), 0, stream>>>(
      w_proj, wprojT, D, D, 0, 0);

  // qkv[b,t,3d] = x[b*t,d] @ wqkvT^T
  gemm_wmma<0><<<dim3(3 * D / 64, Bb * T / 64, 1), 128, 0, stream>>>(
      xb, wqkvT, qkvb, nullptr, D, D, D, 3 * D, 0, 0, 0, 1.0f);

  // V [b,t,d] (strided inside qkv) -> vT [b,d,t]
  transpose_bf16<__bf16><<<dim3(D / 32, T / 32, Bb), dim3(32, 8), 0, stream>>>(
      qkvb + 2 * D, vT, 3 * D, T, (long long)T * 3 * D, (long long)D * T);

  // S[b] = Q[b] @ K[b]^T * (1/32), causal mask -> bf16
  gemm_wmma<1><<<dim3(T / 64, T / 64, Bb), 128, 0, stream>>>(
      qkvb /*Q*/, qkvb + D /*K*/, pb, nullptr, D, 3 * D, 3 * D, T,
      (long long)T * 3 * D, (long long)T * 3 * D, (long long)T * T, 0.03125f);

  softmax_rows<<<dim3(T, Bb), 256, 0, stream>>>(pb, T);

  // out[b] = P[b] @ vT[b]^T  (= P @ V)
  gemm_wmma<0><<<dim3(D / 64, T / 64, Bb), 128, 0, stream>>>(
      pb, vT, outb, nullptr, T, T, T, D,
      (long long)T * T, (long long)D * T, (long long)T * D, 1.0f);

  // y = out[b*t,d] @ wprojT^T + b_proj   (fp32 output)
  gemm_wmma<2><<<dim3(D / 64, Bb * T / 64, 1), 128, 0, stream>>>(
      outb, wprojT, y, b_proj, D, D, D, D, 0, 0, 0, 1.0f);
}
